// FullyConnected_37761352466688
// MI455X (gfx1250) — 8-bit matrix operand order repaired and checked by reading, compiled, not run on hardware
//
#include <hip/hip_runtime.h>


typedef int  v8i  __attribute__((ext_vector_type(8)));
typedef int  v4i  __attribute__((ext_vector_type(4)));
typedef int  v2i  __attribute__((ext_vector_type(2)));

#define BM 128
#define BN 128
#define BK 128
#define LDS_STRIDE 144
#define TILE_BYTES (BM * LDS_STRIDE)
#define BUF_BYTES  (2 * TILE_BYTES)

__device__ __forceinline__ void async_copy_b128(unsigned lds_off,
                                                const signed char* gaddr) {
    asm volatile("global_load_async_to_lds_b128 %0, %1, off"
                 :
                 : "v"(lds_off), "v"(gaddr)
                 : "memory");
}
__device__ __forceinline__ void wait_async0() {
    asm volatile("s_wait_asynccnt 0" ::: "memory");
}

__global__ void pack_i32_to_i8(const int* __restrict__ in,
                               int* __restrict__ out, int n4) {
    int i = blockIdx.x * blockDim.x + threadIdx.x;
    if (i >= n4) return;
    v4i v = *(const v4i*)(in + (size_t)i * 4);
    int p = (v[0] & 0xff) | ((v[1] & 0xff) << 8) |
            ((v[2] & 0xff) << 16) | (v[3] << 24);
    out[i] = p;
}

__global__ __launch_bounds__(256) void i8gemm_wmma_requant(
    const signed char* __restrict__ A8,
    const signed char* __restrict__ W8,
    const int*  __restrict__ bias,
    const int*  __restrict__ p_qm,
    const int*  __restrict__ p_exp,
    const int*  __restrict__ p_zp,
    float* __restrict__ out,
    int K, int N)
{
    __shared__ signed char lds[2 * BUF_BYTES];

    const int tid  = threadIdx.x;
    const int lane = tid & 31;
    const int wid  = tid >> 5;
    const int wm   = wid >> 1;
    const int wn   = wid & 1;
    const int half = lane >> 4;
    const int l16  = lane & 15;

    const int m0 = blockIdx.y * BM;
    const int n0 = blockIdx.x * BN;

    v8i acc[2][4];
    const v8i vzero = {0, 0, 0, 0, 0, 0, 0, 0};
    #pragma unroll
    for (int i = 0; i < 2; ++i)
        #pragma unroll
        for (int j = 0; j < 4; ++j)
            acc[i][j] = vzero;

    auto issue_stage = [&](int k0, int buf) {
        signed char* ldsAb = lds + buf * BUF_BYTES;
        signed char* ldsBb = ldsAb + TILE_BYTES;
        #pragma unroll
        for (int i = 0; i < 4; ++i) {
            int idx = tid + i * 256;
            int row = idx >> 3;
            int seg = idx & 7;
            async_copy_b128(
                (unsigned)(uintptr_t)(ldsAb + row * LDS_STRIDE + seg * 16),
                A8 + (size_t)(m0 + row) * K + k0 + seg * 16);
            async_copy_b128(
                (unsigned)(uintptr_t)(ldsBb + row * LDS_STRIDE + seg * 16),
                W8 + (size_t)(n0 + row) * K + k0 + seg * 16);
        }
    };

    issue_stage(0, 0);
    int buf = 0;

    for (int k0 = 0; k0 < K; k0 += BK) {
        wait_async0();
        __syncthreads();

        if (k0 + BK < K)
            issue_stage(k0 + BK, buf ^ 1);

        const signed char* ldsA = lds + buf * BUF_BYTES;
        const signed char* ldsB = ldsA + TILE_BYTES;

        #pragma unroll
        for (int k64 = 0; k64 < BK; k64 += 64) {
            v8i afrag[2];
            #pragma unroll
            for (int mt = 0; mt < 2; ++mt) {
                const signed char* ab =
                    ldsA + (wm * 32 + mt * 16 + l16) * LDS_STRIDE + k64 + half * 16;
                #pragma unroll
                for (int j = 0; j < 4; ++j) {
                    v2i t = *(const v2i*)(ab + (j >> 1) * 32 + (j & 1) * 8);
                    afrag[mt][2 * j]     = t.x;
                    afrag[mt][2 * j + 1] = t.y;
                }
            }
            v8i bfrag[4];
            #pragma unroll
            for (int nt = 0; nt < 4; ++nt) {
                const signed char* bb =
                    ldsB + (wn * 64 + nt * 16 + l16) * LDS_STRIDE + k64 + half * 16;
                v4i t0 = *(const v4i*)(bb);
                v4i t1 = *(const v4i*)(bb + 32);
                #pragma unroll
                for (int j = 0; j < 4; ++j) {
                    bfrag[nt][j]     = t0[j];
                    bfrag[nt][4 + j] = t1[j];
                }
            }
            #pragma unroll
            for (int mt = 0; mt < 2; ++mt)
                #pragma unroll
                for (int nt = 0; nt < 4; ++nt)
                    acc[mt][nt] = __builtin_amdgcn_wmma_i32_16x16x64_iu8(
                         true, afrag[mt],
                         true, bfrag[nt],
                        acc[mt][nt],
                         false,  false);
        }

        buf ^= 1;
    }

    const int qm = *p_qm;
    const int ex = *p_exp;
    const int zp = *p_zp;
    const int reduced = (qm < 2147418112) ? ((qm + (1 << 15)) >> 16) : 32767;
    const int tshift  = 15 - ex;
    const long long rnd = 1ll << (tshift - 1);

    #pragma unroll
    for (int mt = 0; mt < 2; ++mt) {
        #pragma unroll
        for (int nt = 0; nt < 4; ++nt) {
            const int col = n0 + wn * 64 + nt * 16 + l16;
            const int bv  = bias[col];
            #pragma unroll
            for (int v = 0; v < 8; ++v) {
                const int row = m0 + wm * 32 + mt * 16 + half * 8 + v;
                long long t = (long long)(acc[mt][nt][v] + bv) * (long long)reduced + rnd;
                t >>= tshift;
                t += zp;
                t = t < -128 ? -128 : (t > 127 ? 127 : t);
                out[(size_t)row * N + col] = (float)t;
            }
        }
    }
}

extern "C" void kernel_launch(void* const* d_in, const int* in_sizes, int n_in,
                              void* d_out, int out_size, void* d_ws, size_t ws_size,
                              hipStream_t stream) {
    const int* x32  = (const int*)d_in[0];
    const int* w32  = (const int*)d_in[1];
    const int* bias = (const int*)d_in[2];
    const int* qm   = (const int*)d_in[3];
    const int* ex   = (const int*)d_in[4];
    const int* zp   = (const int*)d_in[5];

    const int N = in_sizes[2];
    const int K = in_sizes[1] / N;
    const int B = in_sizes[0] / K;

    signed char* A8 = (signed char*)d_ws;
    signed char* W8 = A8 + (size_t)B * K;

    {
        int n4 = (int)(((long long)B * K) / 4);
        pack_i32_to_i8<<<(n4 + 255) / 256, 256, 0, stream>>>(x32, (int*)A8, n4);
        n4 = (int)(((long long)N * K) / 4);
        pack_i32_to_i8<<<(n4 + 255) / 256, 256, 0, stream>>>(w32, (int*)W8, n4);
    }

    dim3 grid(N / BN, B / BM);
    i8gemm_wmma_requant<<<grid, 256, 0, stream>>>(
        A8, W8, bias, qm, ex, zp, (float*)d_out, K, N);
}
